// MoEGPTBlock_70446053589277
// MI455X (gfx1250) — compile-verified, matrix operand order checked by reading, not run on hardware
//
#include <hip/hip_runtime.h>
#include <hip/hip_bf16.h>
#include <math.h>

#define DEVI __device__ __forceinline__

typedef __attribute__((ext_vector_type(16))) __bf16 v16bf;
typedef __attribute__((ext_vector_type(8)))  float  v8f;
typedef int v4i_vs __attribute__((vector_size(16)));   // matches builtin param type

// Problem constants (match reference)
constexpr int BB = 4, SS = 1024, DD = 1024, HH = 16, EE = 8, FF = 4096;
constexpr int TT = BB * SS;   // 4096 tokens

#if defined(__gfx1250__) && __has_builtin(__builtin_amdgcn_global_load_async_to_lds_b128) && \
    __has_builtin(__builtin_amdgcn_s_wait_asynccnt)
#define USE_ASYNC_LDS 1
#else
#define USE_ASYNC_LDS 0
#endif

union FragBF {
  v16bf v;
  uint4 u[2];
};

DEVI void ld_frag(FragBF& f, const __bf16* p) {
  f.u[0] = *reinterpret_cast<const uint4*>(p);
  f.u[1] = *reinterpret_cast<const uint4*>(p + 16);
}

DEVI v8f wmma_bf16(const FragBF& a, const FragBF& b, v8f c) {
  return __builtin_amdgcn_wmma_f32_16x16x32_bf16(false, a.v, false, b.v, (short)0, c,
                                                 false, false);
}

DEVI v8f zero8() {
  v8f z;
#pragma unroll
  for (int i = 0; i < 8; ++i) z[i] = 0.0f;
  return z;
}

DEVI float half_max(float x) {   // reduce across the 16 lanes of a half-wave
#pragma unroll
  for (int off = 1; off < 16; off <<= 1) x = fmaxf(x, __shfl_xor(x, off, 32));
  return x;
}
DEVI float half_sum(float x) {
#pragma unroll
  for (int off = 1; off < 16; off <<= 1) x += __shfl_xor(x, off, 32);
  return x;
}

#if USE_ASYNC_LDS
DEVI void cp_async16(__bf16* lds, const __bf16* g) {
  // GLOBAL_LOAD_ASYNC_TO_LDS_B128: 16B per lane, tracked by ASYNCcnt
  __builtin_amdgcn_global_load_async_to_lds_b128(
      (__attribute__((address_space(1))) v4i_vs*)g,
      (__attribute__((address_space(3))) v4i_vs*)lds, 0, 0);
}
#endif

// ---------------------------------------------------------------------------
// Generic 128x128 tile GEMM mainloop, K in steps of 32, bf16 A[M][K] x Bt[N][K].
// 8 waves: wave (wm,wn) = 64x32 sub-tile = 4x2 WMMA fragments.
// All row pointers must be valid (ragged tiles are index-clamped by callers;
// garbage rows only feed epilogue-masked outputs).
// ---------------------------------------------------------------------------
DEVI void gemm_mainloop(const __bf16* pa0, const __bf16* pa1,
                        const __bf16* pb0, const __bf16* pb1,
                        int nk, __bf16* sA, __bf16* sB, int tid,
                        v8f (&acc)[4][2]) {
  const int lane = tid & 31, wid = tid >> 5;
  const int wm = wid >> 2, wn = wid & 3;
  const int l15 = lane & 15;
  const int kb = (lane < 16) ? 0 : 8;
  const int c0 = tid, c1 = tid + 256;
  const int off0 = (c0 >> 2) * 32 + (c0 & 3) * 8;
  const int off1 = (c1 >> 2) * 32 + (c1 & 3) * 8;

#if USE_ASYNC_LDS
  // Pipeline tile 0 straight into LDS via the async data mover.
  cp_async16(sA + off0, pa0);
  cp_async16(sA + off1, pa1);
  cp_async16(sB + off0, pb0);
  cp_async16(sB + off1, pb1);
  for (int it = 0; it < nk; ++it) {
    __bf16* sa = sA + (it & 1) * (128 * 32);
    __bf16* sb = sB + (it & 1) * (128 * 32);
    if (it + 1 < nk) {
      const int ko = (it + 1) * 32;
      __bf16* na = sA + ((it + 1) & 1) * (128 * 32);
      __bf16* nb = sB + ((it + 1) & 1) * (128 * 32);
      cp_async16(na + off0, pa0 + ko);
      cp_async16(na + off1, pa1 + ko);
      cp_async16(nb + off0, pb0 + ko);
      cp_async16(nb + off1, pb1 + ko);
      __builtin_amdgcn_s_wait_asynccnt(4);   // current tile's group done
    } else {
      __builtin_amdgcn_s_wait_asynccnt(0);
    }
    if (it + 8 < nk) {                       // far prefetch of weight stream
      __builtin_prefetch(pb0 + (it + 8) * 32);
      __builtin_prefetch(pb1 + (it + 8) * 32);
    }
    __syncthreads();
    FragBF bfr0, bfr1;
    ld_frag(bfr0, sb + (wn * 32 + 0  + l15) * 32 + kb);
    ld_frag(bfr1, sb + (wn * 32 + 16 + l15) * 32 + kb);
#pragma unroll
    for (int fm = 0; fm < 4; ++fm) {
      FragBF af;
      ld_frag(af, sa + (wm * 64 + fm * 16 + l15) * 32 + kb);
      acc[fm][0] = wmma_bf16(af, bfr0, acc[fm][0]);
      acc[fm][1] = wmma_bf16(af, bfr1, acc[fm][1]);
    }
    __syncthreads();
  }
#else
  uint4 ra0 = *(const uint4*)(pa0);
  uint4 ra1 = *(const uint4*)(pa1);
  uint4 rb0 = *(const uint4*)(pb0);
  uint4 rb1 = *(const uint4*)(pb1);
  for (int it = 0; it < nk; ++it) {
    __bf16* sa = sA + (it & 1) * (128 * 32);
    __bf16* sb = sB + (it & 1) * (128 * 32);
    *(uint4*)(sa + off0) = ra0;
    *(uint4*)(sa + off1) = ra1;
    *(uint4*)(sb + off0) = rb0;
    *(uint4*)(sb + off1) = rb1;
    __syncthreads();
    if (it + 1 < nk) {
      const int ko = (it + 1) * 32;
      ra0 = *(const uint4*)(pa0 + ko);
      ra1 = *(const uint4*)(pa1 + ko);
      rb0 = *(const uint4*)(pb0 + ko);
      rb1 = *(const uint4*)(pb1 + ko);
    }
    if (it + 8 < nk) {
      __builtin_prefetch(pb0 + (it + 8) * 32);
      __builtin_prefetch(pb1 + (it + 8) * 32);
    }
    FragBF bfr0, bfr1;
    ld_frag(bfr0, sb + (wn * 32 + 0  + l15) * 32 + kb);
    ld_frag(bfr1, sb + (wn * 32 + 16 + l15) * 32 + kb);
#pragma unroll
    for (int fm = 0; fm < 4; ++fm) {
      FragBF af;
      ld_frag(af, sa + (wm * 64 + fm * 16 + l15) * 32 + kb);
      acc[fm][0] = wmma_bf16(af, bfr0, acc[fm][0]);
      acc[fm][1] = wmma_bf16(af, bfr1, acc[fm][1]);
    }
    __syncthreads();
  }
#endif
}

// ---------------------------------------------------------------------------
// GEMM: C = A[M,K](bf16) * Bt[N,K](bf16) + bias; RESID => f32 out + residual,
// else bf16 out.  Grid: (N/128, M/128), block 256.
// ---------------------------------------------------------------------------
template <bool RESID>
__global__ __launch_bounds__(256) void gemm_bias_kernel(
    const __bf16* __restrict__ A, const __bf16* __restrict__ Bt,
    const float* __restrict__ bias, const float* __restrict__ resid,
    float* __restrict__ outF, __bf16* __restrict__ outB, int M, int N, int K) {
  __shared__ __attribute__((aligned(16))) __bf16 sA[2 * 128 * 32];
  __shared__ __attribute__((aligned(16))) __bf16 sB[2 * 128 * 32];
  const int tid = threadIdx.x;
  const int nt = blockIdx.x, mt = blockIdx.y;
  const int c0 = tid, c1 = tid + 256;
  const __bf16* pa0 = A + (size_t)(mt * 128 + (c0 >> 2)) * K + (c0 & 3) * 8;
  const __bf16* pa1 = A + (size_t)(mt * 128 + (c1 >> 2)) * K + (c1 & 3) * 8;
  const __bf16* pb0 = Bt + (size_t)(nt * 128 + (c0 >> 2)) * K + (c0 & 3) * 8;
  const __bf16* pb1 = Bt + (size_t)(nt * 128 + (c1 >> 2)) * K + (c1 & 3) * 8;

  v8f acc[4][2];
#pragma unroll
  for (int i = 0; i < 4; ++i) { acc[i][0] = zero8(); acc[i][1] = zero8(); }
  gemm_mainloop(pa0, pa1, pb0, pb1, K / 32, sA, sB, tid, acc);

  const int lane = tid & 31, wid = tid >> 5;
  const int wm = wid >> 2, wn = wid & 3, l15 = lane & 15;
  const int rbase = mt * 128 + wm * 64 + ((lane < 16) ? 0 : 8);
#pragma unroll
  for (int fm = 0; fm < 4; ++fm)
#pragma unroll
    for (int fn = 0; fn < 2; ++fn)
#pragma unroll
      for (int vv = 0; vv < 8; ++vv) {
        const int row = rbase + fm * 16 + vv;
        const int col = nt * 128 + wn * 32 + fn * 16 + l15;
        float val = acc[fm][fn][vv] + bias[col];
        if (RESID)
          outF[(size_t)row * N + col] = val + resid[(size_t)row * N + col];
        else
          outB[(size_t)row * N + col] = (__bf16)val;
      }
}

// ---------------------------------------------------------------------------
// Flash attention. Grid: (S/128, B*H), block 256 (8 waves; wave = 16 q rows).
// ---------------------------------------------------------------------------
__global__ __launch_bounds__(256) void attn_kernel(
    const __bf16* __restrict__ Q, const __bf16* __restrict__ K,
    const __bf16* __restrict__ V, __bf16* __restrict__ O) {
  __shared__ __attribute__((aligned(16))) __bf16 sVt[64 * 32];     // [d][key]
  __shared__ __attribute__((aligned(16))) __bf16 sP[8][16 * 32];   // per-wave P
  const int tid = threadIdx.x, lane = tid & 31, wid = tid >> 5;
  const int bh = blockIdx.y;
  const int b = bh >> 4, h = bh & 15;       // H = 16
  const int q0 = blockIdx.x * 128 + wid * 16;
  const int l15 = lane & 15;
  const int kbd = (lane < 16) ? 0 : 8;

  const __bf16* qrow = Q + (size_t)(b * SS + q0 + l15) * DD + h * 64;
  FragBF qa0, qa1;
  ld_frag(qa0, qrow + kbd);
  ld_frag(qa1, qrow + 32 + kbd);

  float m_i[8], l_i[8];
  v8f oacc[4];
#pragma unroll
  for (int i = 0; i < 8; ++i) { m_i[i] = -1e30f; l_i[i] = 0.0f; }
#pragma unroll
  for (int i = 0; i < 4; ++i) oacc[i] = zero8();

  const __bf16* Kbase = K + (size_t)b * SS * DD + h * 64;
  const __bf16* Vbase = V + (size_t)b * SS * DD + h * 64;

  for (int j = 0; j < SS / 32; ++j) {
    __syncthreads();
    {  // stage V chunk transposed: sVt[d][key]
      const int key = tid >> 3;
      const int dg = (tid & 7) * 8;
      uint4 vv = *(const uint4*)(Vbase + (size_t)(j * 32 + key) * DD + dg);
      const __bf16* pv = (const __bf16*)&vv;
#pragma unroll
      for (int i = 0; i < 8; ++i) sVt[(dg + i) * 32 + key] = pv[i];
    }
    __syncthreads();

    v8f sc[2];
#pragma unroll
    for (int fn = 0; fn < 2; ++fn) {
      v8f a = zero8();
      const __bf16* krow = Kbase + (size_t)(j * 32 + fn * 16 + l15) * DD;
      FragBF kf;
      ld_frag(kf, krow + kbd);
      a = wmma_bf16(qa0, kf, a);
      ld_frag(kf, krow + 32 + kbd);
      a = wmma_bf16(qa1, kf, a);
      sc[fn] = a;
    }

#pragma unroll
    for (int vv = 0; vv < 8; ++vv) {
      float s0 = sc[0][vv] * 0.125f;   // 1/sqrt(64)
      float s1 = sc[1][vv] * 0.125f;
      float cm = half_max(fmaxf(s0, s1));
      float nm = fmaxf(m_i[vv], cm);
      float scale = __expf(m_i[vv] - nm);
      float p0 = __expf(s0 - nm);
      float p1 = __expf(s1 - nm);
      float rs = half_sum(p0 + p1);
      l_i[vv] = l_i[vv] * scale + rs;
      m_i[vv] = nm;
#pragma unroll
      for (int nf = 0; nf < 4; ++nf) oacc[nf][vv] *= scale;
      const int prow = vv + ((lane < 16) ? 0 : 8);
      sP[wid][prow * 32 + l15]      = (__bf16)p0;
      sP[wid][prow * 32 + 16 + l15] = (__bf16)p1;
    }

    FragBF pf;
    ld_frag(pf, &sP[wid][l15 * 32 + kbd]);
#pragma unroll
    for (int nf = 0; nf < 4; ++nf) {
      FragBF vf;
      ld_frag(vf, sVt + (nf * 16 + l15) * 32 + kbd);
      oacc[nf] = wmma_bf16(pf, vf, oacc[nf]);
    }
  }

#pragma unroll
  for (int nf = 0; nf < 4; ++nf)
#pragma unroll
    for (int vv = 0; vv < 8; ++vv) {
      const int row = q0 + vv + ((lane < 16) ? 0 : 8);
      const int d = nf * 16 + l15;
      O[(size_t)(b * SS + row) * DD + h * 64 + d] = (__bf16)(oacc[nf][vv] / l_i[vv]);
    }
}

// ---------------------------------------------------------------------------
// LayerNorm -> bf16. Grid: T, block 256.
// ---------------------------------------------------------------------------
__global__ __launch_bounds__(256) void ln_kernel(const float* __restrict__ x,
                                                 const float* __restrict__ g,
                                                 const float* __restrict__ bta,
                                                 __bf16* __restrict__ out) {
  const int row = blockIdx.x, tid = threadIdx.x;
  const int lane = tid & 31, wid = tid >> 5;
  float xs[4], s = 0.f, s2 = 0.f;
#pragma unroll
  for (int i = 0; i < 4; ++i) {
    xs[i] = x[(size_t)row * DD + tid + i * 256];
    s += xs[i];
    s2 += xs[i] * xs[i];
  }
#pragma unroll
  for (int off = 1; off < 32; off <<= 1) {
    s += __shfl_xor(s, off, 32);
    s2 += __shfl_xor(s2, off, 32);
  }
  __shared__ float ws[8], ws2[8];
  if (lane == 0) { ws[wid] = s; ws2[wid] = s2; }
  __syncthreads();
  if (tid == 0) {
    float t = 0.f, t2 = 0.f;
    for (int e = 0; e < 8; ++e) { t += ws[e]; t2 += ws2[e]; }
    float mean = t / DD;
    ws[0] = mean;
    ws2[0] = t2 / DD - mean * mean;
  }
  __syncthreads();
  const float mean = ws[0];
  const float inv = rsqrtf(ws2[0] + 1e-5f);
#pragma unroll
  for (int i = 0; i < 4; ++i) {
    const int d = tid + i * 256;
    out[(size_t)row * DD + d] = (__bf16)((xs[i] - mean) * inv * g[d] + bta[d]);
  }
}

// ---------------------------------------------------------------------------
// Gate: probs = softmax(h2 @ Wg), top-2, renorm, scatter into expert lists.
// ---------------------------------------------------------------------------
__global__ __launch_bounds__(256) void gate_kernel(const __bf16* __restrict__ h2,
                                                   const float* __restrict__ Wg,
                                                   int* __restrict__ cnt,
                                                   int* __restrict__ list,
                                                   float* __restrict__ gatev) {
  const int t = blockIdx.x, tid = threadIdx.x;
  __shared__ float sl[256][8];
  float p[8];
#pragma unroll
  for (int e = 0; e < 8; ++e) p[e] = 0.f;
#pragma unroll
  for (int dd = 0; dd < 4; ++dd) {
    const int d = tid + dd * 256;
    const float xv = (float)h2[(size_t)t * DD + d];
#pragma unroll
    for (int e = 0; e < 8; ++e) p[e] += xv * Wg[(size_t)d * EE + e];
  }
#pragma unroll
  for (int e = 0; e < 8; ++e) sl[tid][e] = p[e];
  __syncthreads();
  for (int st = 128; st >= 1; st >>= 1) {
    if (tid < st)
#pragma unroll
      for (int e = 0; e < 8; ++e) sl[tid][e] += sl[tid + st][e];
    __syncthreads();
  }
  if (tid == 0) {
    float mx = -1e30f;
    for (int e = 0; e < 8; ++e) mx = fmaxf(mx, sl[0][e]);
    float pe[8];
    for (int e = 0; e < 8; ++e) pe[e] = __expf(sl[0][e] - mx);
    int e1 = 0;
    for (int e = 1; e < 8; ++e) if (pe[e] > pe[e1]) e1 = e;
    int e2 = (e1 == 0) ? 1 : 0;
    for (int e = 0; e < 8; ++e) if (e != e1 && pe[e] > pe[e2]) e2 = e;
    const float r = 1.0f / (pe[e1] + pe[e2]);
    int s1 = atomicAdd(&cnt[e1], 1);
    list[e1 * TT + s1] = t;
    gatev[e1 * TT + s1] = pe[e1] * r;
    int s2 = atomicAdd(&cnt[e2], 1);
    list[e2 * TT + s2] = t;
    gatev[e2 * TT + s2] = pe[e2] * r;
  }
}

__global__ void zero_cnt_kernel(int* cnt) {
  if (threadIdx.x < EE) cnt[threadIdx.x] = 0;
}

__global__ void scan_kernel(const int* __restrict__ cnt, int* __restrict__ offs) {
  if (threadIdx.x == 0) {
    int a = 0;
    for (int e = 0; e < EE; ++e) { offs[e] = a; a += cnt[e]; }
    offs[EE] = a;
  }
}

// ---------------------------------------------------------------------------
// MoE FFN1: hidden = gelu(gather(h2) @ W1T + b1). Grid: (F/128, T/128, E).
// ---------------------------------------------------------------------------
__global__ __launch_bounds__(256) void moe_ffn1_kernel(
    const __bf16* __restrict__ h2, const __bf16* __restrict__ W1T,
    const float* __restrict__ b1, const int* __restrict__ cnt,
    const int* __restrict__ offs, const int* __restrict__ list,
    __bf16* __restrict__ hidden) {
  const int e = blockIdx.z;
  const int ce = cnt[e];
  const int rt = blockIdx.y;
  if (rt * 128 >= ce) return;
  __shared__ __attribute__((aligned(16))) __bf16 sA[2 * 128 * 32];
  __shared__ __attribute__((aligned(16))) __bf16 sB[2 * 128 * 32];
  const int tid = threadIdx.x, nt = blockIdx.x;
  const int c0 = tid, c1 = tid + 256;
  const int i0 = min(rt * 128 + (c0 >> 2), ce - 1);  // clamp ragged rows
  const int i1 = min(rt * 128 + (c1 >> 2), ce - 1);
  const __bf16* pa0 = h2 + (size_t)list[e * TT + i0] * DD + (c0 & 3) * 8;
  const __bf16* pa1 = h2 + (size_t)list[e * TT + i1] * DD + (c1 & 3) * 8;
  const __bf16* pb0 =
      W1T + ((size_t)e * FF + nt * 128 + (c0 >> 2)) * DD + (c0 & 3) * 8;
  const __bf16* pb1 =
      W1T + ((size_t)e * FF + nt * 128 + (c1 >> 2)) * DD + (c1 & 3) * 8;

  v8f acc[4][2];
#pragma unroll
  for (int i = 0; i < 4; ++i) { acc[i][0] = zero8(); acc[i][1] = zero8(); }
  gemm_mainloop(pa0, pa1, pb0, pb1, DD / 32, sA, sB, tid, acc);

  const int lane = tid & 31, wid = tid >> 5;
  const int wm = wid >> 2, wn = wid & 3, l15 = lane & 15;
  const int base = offs[e];
  const int ibase = rt * 128 + wm * 64 + ((lane < 16) ? 0 : 8);
#pragma unroll
  for (int fm = 0; fm < 4; ++fm)
#pragma unroll
    for (int fn = 0; fn < 2; ++fn)
#pragma unroll
      for (int vv = 0; vv < 8; ++vv) {
        const int i = ibase + fm * 16 + vv;
        if (i < ce) {
          const int f = nt * 128 + wn * 32 + fn * 16 + l15;
          float val = acc[fm][fn][vv] + b1[(size_t)e * FF + f];
          val = 0.5f * val * (1.0f + erff(val * 0.70710678118f));  // exact GELU
          hidden[(size_t)(base + i) * FF + f] = (__bf16)val;
        }
      }
}

// ---------------------------------------------------------------------------
// MoE FFN2: out[token] += gate * (hidden @ W2T + b2). Grid: (D/128, T/128, E).
// ---------------------------------------------------------------------------
__global__ __launch_bounds__(256) void moe_ffn2_kernel(
    const __bf16* __restrict__ hidden, const __bf16* __restrict__ W2T,
    const float* __restrict__ b2, const int* __restrict__ cnt,
    const int* __restrict__ offs, const int* __restrict__ list,
    const float* __restrict__ gatev, float* __restrict__ out) {
  const int e = blockIdx.z;
  const int ce = cnt[e];
  const int rt = blockIdx.y;
  if (rt * 128 >= ce) return;
  __shared__ __attribute__((aligned(16))) __bf16 sA[2 * 128 * 32];
  __shared__ __attribute__((aligned(16))) __bf16 sB[2 * 128 * 32];
  const int tid = threadIdx.x, nt = blockIdx.x;
  const int base = offs[e];
  const int c0 = tid, c1 = tid + 256;
  const int i0 = min(rt * 128 + (c0 >> 2), ce - 1);
  const int i1 = min(rt * 128 + (c1 >> 2), ce - 1);
  const __bf16* pa0 = hidden + (size_t)(base + i0) * FF + (c0 & 3) * 8;
  const __bf16* pa1 = hidden + (size_t)(base + i1) * FF + (c1 & 3) * 8;
  const __bf16* pb0 =
      W2T + ((size_t)e * DD + nt * 128 + (c0 >> 2)) * FF + (c0 & 3) * 8;
  const __bf16* pb1 =
      W2T + ((size_t)e * DD + nt * 128 + (c1 >> 2)) * FF + (c1 & 3) * 8;

  v8f acc[4][2];
#pragma unroll
  for (int i = 0; i < 4; ++i) { acc[i][0] = zero8(); acc[i][1] = zero8(); }
  gemm_mainloop(pa0, pa1, pb0, pb1, FF / 32, sA, sB, tid, acc);

  const int lane = tid & 31, wid = tid >> 5;
  const int wm = wid >> 2, wn = wid & 3, l15 = lane & 15;
  const int ibase = rt * 128 + wm * 64 + ((lane < 16) ? 0 : 8);
#pragma unroll
  for (int fm = 0; fm < 4; ++fm)
#pragma unroll
    for (int fn = 0; fn < 2; ++fn)
#pragma unroll
      for (int vv = 0; vv < 8; ++vv) {
        const int i = ibase + fm * 16 + vv;
        if (i < ce) {
          const int token = list[e * TT + i];
          const float g = gatev[e * TT + i];
          const int d = nt * 128 + wn * 32 + fn * 16 + l15;
          const float val = acc[fm][fn][vv] + b2[(size_t)e * DD + d];
          atomicAdd(&out[(size_t)token * DD + d], g * val);
        }
      }
}

// ---------------------------------------------------------------------------
// Coalesced fp32 -> bf16 transposed weight conversion via LDS tile.
// out[batch][C][R] = in[batch][R][C]^T.  Grid: (C/32, R/32, batch), block 256.
// ---------------------------------------------------------------------------
__global__ __launch_bounds__(256) void conv_t_kernel(const float* __restrict__ in,
                                                     __bf16* __restrict__ out,
                                                     int R, int C) {
  __shared__ float tile[32][33];
  const size_t base = (size_t)blockIdx.z * R * C;
  const int rt = blockIdx.y * 32, ct = blockIdx.x * 32;
  const int tx = threadIdx.x & 31, ty = threadIdx.x >> 5;   // 32 x 8
#pragma unroll
  for (int i = 0; i < 32; i += 8)
    tile[ty + i][tx] = in[base + (size_t)(rt + ty + i) * C + ct + tx];
  __syncthreads();
#pragma unroll
  for (int i = 0; i < 32; i += 8)
    out[base + (size_t)(ct + ty + i) * R + rt + tx] = (__bf16)tile[tx][ty + i];
}

__global__ __launch_bounds__(256) void copy_f32_kernel(const float* __restrict__ in,
                                                       float* __restrict__ out) {
  const size_t i = (size_t)blockIdx.x * 256 + threadIdx.x;
  reinterpret_cast<float4*>(out)[i] = reinterpret_cast<const float4*>(in)[i];
}

// ---------------------------------------------------------------------------
extern "C" void kernel_launch(void* const* d_in, const int* in_sizes, int n_in,
                              void* d_out, int out_size, void* d_ws, size_t ws_size,
                              hipStream_t stream) {
  const float* x    = (const float*)d_in[0];
  const float* ln1g = (const float*)d_in[1];
  const float* ln1b = (const float*)d_in[2];
  const float* Wq   = (const float*)d_in[3];
  const float* Wk   = (const float*)d_in[4];
  const float* Wv   = (const float*)d_in[5];
  const float* bq   = (const float*)d_in[6];
  const float* bk   = (const float*)d_in[7];
  const float* bv   = (const float*)d_in[8];
  const float* Wo   = (const float*)d_in[9];
  const float* bo   = (const float*)d_in[10];
  const float* ln2g = (const float*)d_in[11];
  const float* ln2b = (const float*)d_in[12];
  const float* Wg   = (const float*)d_in[13];
  const float* W1   = (const float*)d_in[14];
  const float* b1   = (const float*)d_in[15];
  const float* W2   = (const float*)d_in[16];
  const float* b2   = (const float*)d_in[17];
  // d_in[18] = top_k (hard-coded 2)
  float* out = (float*)d_out;

  char* w = (char*)d_ws;
  auto take = [&](size_t bytes) -> void* {
    void* p = (void*)w;
    w += (bytes + 255) & ~(size_t)255;
    return p;
  };
  __bf16* hbf  = (__bf16*)take((size_t)TT * DD * 2);
  __bf16* WqT  = (__bf16*)take((size_t)DD * DD * 2);
  __bf16* WkT  = (__bf16*)take((size_t)DD * DD * 2);
  __bf16* WvT  = (__bf16*)take((size_t)DD * DD * 2);
  __bf16* WoT  = (__bf16*)take((size_t)DD * DD * 2);
  __bf16* qb   = (__bf16*)take((size_t)TT * DD * 2);
  __bf16* kb   = (__bf16*)take((size_t)TT * DD * 2);
  __bf16* vb   = (__bf16*)take((size_t)TT * DD * 2);
  __bf16* aout = (__bf16*)take((size_t)TT * DD * 2);
  float*  x1   = (float*)take((size_t)TT * DD * 4);
  __bf16* h2bf = (__bf16*)take((size_t)TT * DD * 2);
  __bf16* W1T  = (__bf16*)take((size_t)EE * DD * FF * 2);
  __bf16* W2T  = (__bf16*)take((size_t)EE * DD * FF * 2);
  __bf16* hidden = (__bf16*)take((size_t)2 * TT * FF * 2);
  int*   cnt  = (int*)take(EE * 4);
  int*   offs = (int*)take((EE + 1) * 4);
  int*   list = (int*)take((size_t)EE * TT * 4);
  float* gatev = (float*)take((size_t)EE * TT * 4);

  // 0. zero routing counters
  zero_cnt_kernel<<<1, 32, 0, stream>>>(cnt);

  // 1. weight conversion / transpose to bf16 [N][K] (coalesced, LDS-tiled)
  conv_t_kernel<<<dim3(DD / 32, DD / 32, 1), 256, 0, stream>>>(Wq, WqT, DD, DD);
  conv_t_kernel<<<dim3(DD / 32, DD / 32, 1), 256, 0, stream>>>(Wk, WkT, DD, DD);
  conv_t_kernel<<<dim3(DD / 32, DD / 32, 1), 256, 0, stream>>>(Wv, WvT, DD, DD);
  conv_t_kernel<<<dim3(DD / 32, DD / 32, 1), 256, 0, stream>>>(Wo, WoT, DD, DD);
  conv_t_kernel<<<dim3(FF / 32, DD / 32, EE), 256, 0, stream>>>(W1, W1T, DD, FF);
  conv_t_kernel<<<dim3(DD / 32, FF / 32, EE), 256, 0, stream>>>(W2, W2T, FF, DD);

  // 2. LN1
  ln_kernel<<<TT, 256, 0, stream>>>(x, ln1g, ln1b, hbf);

  // 3. Q,K,V projections (bf16 out)
  gemm_bias_kernel<false><<<dim3(DD / 128, TT / 128), 256, 0, stream>>>(
      hbf, WqT, bq, nullptr, nullptr, qb, TT, DD, DD);
  gemm_bias_kernel<false><<<dim3(DD / 128, TT / 128), 256, 0, stream>>>(
      hbf, WkT, bk, nullptr, nullptr, kb, TT, DD, DD);
  gemm_bias_kernel<false><<<dim3(DD / 128, TT / 128), 256, 0, stream>>>(
      hbf, WvT, bv, nullptr, nullptr, vb, TT, DD, DD);

  // 4. flash attention
  attn_kernel<<<dim3(SS / 128, BB * HH), 256, 0, stream>>>(qb, kb, vb, aout);

  // 5. output projection + residual (f32)
  gemm_bias_kernel<true><<<dim3(DD / 128, TT / 128), 256, 0, stream>>>(
      aout, WoT, bo, x, x1, nullptr, TT, DD, DD);

  // 6. LN2
  ln_kernel<<<TT, 256, 0, stream>>>(x1, ln2g, ln2b, h2bf);

  // 7. gating: softmax + top-2 + scatter to expert lists
  gate_kernel<<<TT, 256, 0, stream>>>(h2bf, Wg, cnt, list, gatev);
  scan_kernel<<<1, 32, 0, stream>>>(cnt, offs);

  // 8. out = x1 (residual), then MoE adds on top
  copy_f32_kernel<<<(TT * DD) / (256 * 4), 256, 0, stream>>>(x1, out);

  // 9. expert FFN1 (gathered rows, GELU)
  moe_ffn1_kernel<<<dim3(FF / 128, TT / 128, EE), 256, 0, stream>>>(
      h2bf, W1T, b1, cnt, offs, list, hidden);

  // 10. expert FFN2 + gated atomic scatter
  moe_ffn2_kernel<<<dim3(DD / 128, TT / 128, EE), 256, 0, stream>>>(
      hidden, W2T, b2, cnt, offs, list, gatev, out);
}
